// AxialTimeAttention_19954418057505
// MI455X (gfx1250) — matrix operand order repaired and checked by reading, not compiled, not run on hardware
//
#include <hip/hip_runtime.h>

typedef unsigned short u16;
typedef __attribute__((ext_vector_type(16))) __bf16 v16bf;
typedef __attribute__((ext_vector_type(8)))  float  v8f;
typedef __attribute__((ext_vector_type(2)))  float  v2f;

union FragB { v16bf v; uint4 u[2]; };

__device__ __forceinline__ v8f zero8() {
  v8f z;
#pragma unroll
  for (int i = 0; i < 8; ++i) z[i] = 0.f;
  return z;
}

__device__ __forceinline__ u16 f2bf(float f) {
  unsigned int u = __float_as_uint(f);
  u = (u + 0x7FFFu + ((u >> 16) & 1u)) >> 16;
  return (u16)u;
}

// ---------------- constants ----------------
// x: (T=16, B=1, C=768, H=16, W=16, D=8); S = H*W*D = 2048; tokens = T*S = 32768
constexpr int CH    = 768;
constexpr int TT    = 16;
constexpr int SP    = 2048;
constexpr int NTOK  = TT * SP;        // 32768
constexpr int HEADS = 12;
constexpr int HD    = 64;
constexpr int M_QKV = 3 * CH;         // 2304
constexpr float EPSN = 1e-6f;

// ---------------- kernel 1: group RMS stats ----------------
__global__ __launch_bounds__(256) void gn_stats_kernel(const float* __restrict__ x,
                                                       float* __restrict__ rstd) {
  __shared__ float red[256];
  int t = blockIdx.x / HEADS, g = blockIdx.x % HEADS;
  const float* p = x + ((size_t)t * CH + (size_t)g * 64) * SP;
  float acc = 0.f;
  for (int i = threadIdx.x; i < 64 * SP; i += 256) { float v = p[i]; acc += v * v; }
  red[threadIdx.x] = acc;
  __syncthreads();
  for (int s2 = 128; s2 > 0; s2 >>= 1) {
    if ((int)threadIdx.x < s2) red[threadIdx.x] += red[threadIdx.x + s2];
    __syncthreads();
  }
  if (threadIdx.x == 0)
    rstd[blockIdx.x] = rsqrtf(red[0] * (1.f / (64.f * (float)SP)) + EPSN);
}

// ---------------- kernel 2: normalize + transpose (C,S)->(S,C) + bf16 ----------------
__global__ __launch_bounds__(256) void norm_transpose_kernel(const float* __restrict__ x,
                                                             const float* __restrict__ rstd,
                                                             const float* __restrict__ w,
                                                             u16* __restrict__ xbt) {
  __shared__ float tile[32][33];
  int t = blockIdx.z;
  int cbase = blockIdx.y * 32;
  int sbase = blockIdx.x * 32;
  int tx = threadIdx.x, ty = threadIdx.y;   // 32 x 8
#pragma unroll
  for (int i = 0; i < 4; ++i) {
    int cl = ty + i * 8;
    int c = cbase + cl;
    float v = x[((size_t)t * CH + c) * SP + sbase + tx];
    tile[cl][tx] = v * rstd[t * HEADS + (c >> 6)] * w[c];
  }
  __syncthreads();
#pragma unroll
  for (int i = 0; i < 4; ++i) {
    int sl = ty + i * 8;
    xbt[((size_t)t * SP + sbase + sl) * CH + cbase + tx] = f2bf(tile[tx][sl]);
  }
}

// ---------------- kernel 3: fp32 -> bf16 flat convert ----------------
__global__ void cvt_bf16_kernel(const float* __restrict__ in, u16* __restrict__ out, int n) {
  int i = blockIdx.x * 256 + threadIdx.x;
  if (i < n) out[i] = f2bf(in[i]);
}

// ---------------- kernel 4: relative-position bias table ----------------
__global__ void bias_table_kernel(const float* __restrict__ table, float* __restrict__ bm) {
  int idx = blockIdx.x * 256 + threadIdx.x;   // 12*16*16 = 3072
  if (idx >= HEADS * TT * TT) return;
  int h = idx / (TT * TT);
  int rem = idx % (TT * TT);
  int tq = rem / TT, tk = rem % TT;
  int rp = tk - tq;                  // mem - ctx
  int ret = (rp > 0) ? 16 : 0;       // NUM_BUCKETS/2
  int n = rp < 0 ? -rp : rp;
  int b;
  if (n < 8) {                       // max_exact = 8
    b = n;
  } else {
    int v = 8 + (int)(logf((float)n * (1.f / 8.f)) * (1.f / logf(16.f)) * 8.f);
    b = v < 15 ? v : 15;
  }
  bm[idx] = table[(ret + b) * HEADS + h];
}

// ---------------- WMMA helpers ----------------
__device__ __forceinline__ v8f wmma_bf(const FragB& a, const FragB& b, v8f c) {
  return __builtin_amdgcn_wmma_f32_16x16x32_bf16(false, a.v, false, b.v, (short)0, c, false, false);
}
__device__ __forceinline__ v8f wmma_f32(v2f a, v2f b, v8f c) {
  return __builtin_amdgcn_wmma_f32_16x16x4_f32(false, a, false, b, (short)0, c, false, false);
}

// ---------------- kernel 5: QKV GEMM (bf16 WMMA) ----------------
// A: weights (2304 x 768) bf16 row-major; B: tokens (32768 x 768) bf16 token-major.
// Epilogue scatters into qkv[part][head][s][t][d] fp32 (+bias).
__global__ __launch_bounds__(256) void gemm_qkv_kernel(const u16* __restrict__ A,
                                                       const u16* __restrict__ Bm,
                                                       const float* __restrict__ bias,
                                                       float* __restrict__ qkv) {
  const int K = CH;
  int lane = threadIdx.x & 31;
  int wave = threadIdx.x >> 5;
  int tile = blockIdx.x * 8 + wave;
  const int Mt = M_QKV / 32;                 // 72
  int mbase = (tile % Mt) * 32;
  int nbase = (tile / Mt) * 32;
  int half = lane >> 4, r = lane & 15;

  const u16* ap0 = A  + (size_t)(mbase + r) * K;
  const u16* ap1 = ap0 + (size_t)16 * K;
  const u16* bp0 = Bm + (size_t)(nbase + r) * K;
  const u16* bp1 = bp0 + (size_t)16 * K;

  v8f acc[2][2];
  acc[0][0] = zero8(); acc[0][1] = zero8(); acc[1][0] = zero8(); acc[1][1] = zero8();

  for (int kk = 0; kk < K; kk += 32) {
    int ka0 = kk + half * 8;        // A: lane<16 -> K {0..7,16..23}
    int ka1 = kk + 16 + half * 8;
    int kb0 = kk + half * 8;        // B: lane<16 -> K {0..7,16..23}
    FragB a0, a1, b0, b1;
    a0.u[0] = *(const uint4*)(ap0 + ka0); a0.u[1] = *(const uint4*)(ap0 + ka1);
    a1.u[0] = *(const uint4*)(ap1 + ka0); a1.u[1] = *(const uint4*)(ap1 + ka1);
    b0.u[0] = *(const uint4*)(bp0 + kb0); b0.u[1] = *(const uint4*)(bp0 + kb0 + 16);
    b1.u[0] = *(const uint4*)(bp1 + kb0); b1.u[1] = *(const uint4*)(bp1 + kb0 + 16);
    acc[0][0] = wmma_bf(a0, b0, acc[0][0]);
    acc[0][1] = wmma_bf(a0, b1, acc[0][1]);
    acc[1][0] = wmma_bf(a1, b0, acc[1][0]);
    acc[1][1] = wmma_bf(a1, b1, acc[1][1]);
  }

#pragma unroll
  for (int i = 0; i < 2; ++i)
#pragma unroll
    for (int j = 0; j < 2; ++j)
#pragma unroll
      for (int e = 0; e < 8; ++e) {
        int m = mbase + i * 16 + e + 8 * half;
        int n = nbase + j * 16 + r;              // token
        float val = acc[i][j][e] + bias[m];
        int h = m / 192, rr = m % 192;
        int part = rr >> 6, d = rr & 63;
        int t = n >> 11, s = n & 2047;
        size_t di = ((((size_t)part * HEADS + h) * SP + s) * TT + t) * HD + d;
        qkv[di] = val;
      }
}

// ---------------- kernel 6: attention (fp32 WMMA 16x16x4) ----------------
// One wave per (head, s). LN(q,k) in regs -> LDS; S=QK^T via wmma; softmax via
// lane shuffles; P in LDS; O=P@V via wmma; bf16 token-major output (NTOK x 768).
__global__ __launch_bounds__(128) void attn_kernel(const float* __restrict__ qkv,
                                                   const float* __restrict__ qsc,
                                                   const float* __restrict__ qbi,
                                                   const float* __restrict__ ksc,
                                                   const float* __restrict__ kbi,
                                                   const float* __restrict__ biasmat,
                                                   u16* __restrict__ obt) {
  __shared__ float sq[4][TT * HD];
  __shared__ float sk[4][TT * HD];
  __shared__ float sp[4][TT * TT];

  int wave = threadIdx.x >> 5;
  int lane = threadIdx.x & 31;
  int idx = blockIdx.x * 4 + wave;
  int h = idx >> 11;           // idx / SP
  int s = idx & 2047;

  const float* Qg = qkv + (((size_t)(0 * HEADS + h) * SP + s) * TT) * HD;
  const float* Kg = qkv + (((size_t)(1 * HEADS + h) * SP + s) * TT) * HD;
  const float* Vg = qkv + (((size_t)(2 * HEADS + h) * SP + s) * TT) * HD;

  // LayerNorm over HD: two lanes per (t) row.
  int row = lane >> 1;
  int dbase = (lane & 1) * 32;
  {
    const float* qr = Qg + row * HD;
    float vals[32], sum = 0.f, ss = 0.f;
#pragma unroll
    for (int i = 0; i < 32; ++i) { float v = qr[dbase + i]; vals[i] = v; sum += v; ss += v * v; }
    sum += __shfl_xor(sum, 1, 32);
    ss  += __shfl_xor(ss, 1, 32);
    float mu = sum * (1.f / 64.f);
    float inv = rsqrtf(ss * (1.f / 64.f) - mu * mu + EPSN);
#pragma unroll
    for (int i = 0; i < 32; ++i) {
      int d = dbase + i;
      sq[wave][row * HD + d] = (vals[i] - mu) * inv * qsc[d] + qbi[d];
    }
  }
  {
    const float* kr = Kg + row * HD;
    float vals[32], sum = 0.f, ss = 0.f;
#pragma unroll
    for (int i = 0; i < 32; ++i) { float v = kr[dbase + i]; vals[i] = v; sum += v; ss += v * v; }
    sum += __shfl_xor(sum, 1, 32);
    ss  += __shfl_xor(ss, 1, 32);
    float mu = sum * (1.f / 64.f);
    float inv = rsqrtf(ss * (1.f / 64.f) - mu * mu + EPSN);
#pragma unroll
    for (int i = 0; i < 32; ++i) {
      int d = dbase + i;
      sk[wave][row * HD + d] = (vals[i] - mu) * inv * ksc[d] + kbi[d];
    }
  }
  __syncthreads();

  int half = lane >> 4, r = lane & 15;

  // S = Q K^T  (M=tq, N=tk, K=HD)
  v8f accS = zero8();
  for (int kk = 0; kk < HD; kk += 4) {
    int k = kk + half * 2;
    v2f a, b;
    a.x = sq[wave][r * HD + k];  a.y = sq[wave][r * HD + k + 1];
    b.x = sk[wave][r * HD + k];  b.y = sk[wave][r * HD + k + 1];
    accS = wmma_f32(a, b, accS);
  }

  // softmax over tk (16 lanes per half-wave hold one row per acc element)
  const float* bm = biasmat + h * (TT * TT);
  const float scale = 0.125f;  // 1/sqrt(64)
#pragma unroll
  for (int e = 0; e < 8; ++e) {
    int tq = e + 8 * half;
    float sv = accS[e] * scale + bm[tq * TT + r];
    float mx = sv;
    for (int off = 8; off >= 1; off >>= 1) mx = fmaxf(mx, __shfl_xor(mx, off, 16));
    float ev = __expf(sv - mx);
    float sm = ev;
    for (int off = 8; off >= 1; off >>= 1) sm += __shfl_xor(sm, off, 16);
    sp[wave][tq * TT + r] = ev / sm;
  }
  __syncthreads();

  // O = P @ V  (M=tq, K=tk=16, N=HD in 4 chunks of 16)
  for (int nc = 0; nc < 4; ++nc) {
    v8f acco = zero8();
    int col = nc * 16 + r;
    for (int kk = 0; kk < TT; kk += 4) {
      int k = kk + half * 2;
      v2f a, b;
      a.x = sp[wave][r * TT + k];  a.y = sp[wave][r * TT + k + 1];
      b.x = Vg[k * HD + col];      b.y = Vg[(k + 1) * HD + col];
      acco = wmma_f32(a, b, acco);
    }
#pragma unroll
    for (int e = 0; e < 8; ++e) {
      int tq = e + 8 * half;
      size_t token = (size_t)tq * SP + s;
      obt[token * CH + h * HD + col] = f2bf(acco[e]);
    }
  }
}

// ---------------- kernel 7: output GEMM (bf16 WMMA) + bias + residual ----------------
__global__ __launch_bounds__(256) void gemm_out_kernel(const u16* __restrict__ A,
                                                       const u16* __restrict__ Bm,
                                                       const float* __restrict__ bias,
                                                       const float* __restrict__ xres,
                                                       float* __restrict__ y) {
  const int K = CH;
  int lane = threadIdx.x & 31;
  int wave = threadIdx.x >> 5;
  int tile = blockIdx.x * 8 + wave;
  const int Mt = CH / 32;                    // 24
  int mbase = (tile % Mt) * 32;
  int nbase = (tile / Mt) * 32;
  int half = lane >> 4, r = lane & 15;

  const u16* ap0 = A  + (size_t)(mbase + r) * K;
  const u16* ap1 = ap0 + (size_t)16 * K;
  const u16* bp0 = Bm + (size_t)(nbase + r) * K;
  const u16* bp1 = bp0 + (size_t)16 * K;

  v8f acc[2][2];
  acc[0][0] = zero8(); acc[0][1] = zero8(); acc[1][0] = zero8(); acc[1][1] = zero8();

  for (int kk = 0; kk < K; kk += 32) {
    int ka0 = kk + half * 8;
    int ka1 = kk + 16 + half * 8;
    int kb0 = kk + half * 8;
    FragB a0, a1, b0, b1;
    a0.u[0] = *(const uint4*)(ap0 + ka0); a0.u[1] = *(const uint4*)(ap0 + ka1);
    a1.u[0] = *(const uint4*)(ap1 + ka0); a1.u[1] = *(const uint4*)(ap1 + ka1);
    b0.u[0] = *(const uint4*)(bp0 + kb0); b0.u[1] = *(const uint4*)(bp0 + kb0 + 16);
    b1.u[0] = *(const uint4*)(bp1 + kb0); b1.u[1] = *(const uint4*)(bp1 + kb0 + 16);
    acc[0][0] = wmma_bf(a0, b0, acc[0][0]);
    acc[0][1] = wmma_bf(a0, b1, acc[0][1]);
    acc[1][0] = wmma_bf(a1, b0, acc[1][0]);
    acc[1][1] = wmma_bf(a1, b1, acc[1][1]);
  }

#pragma unroll
  for (int i = 0; i < 2; ++i)
#pragma unroll
    for (int j = 0; j < 2; ++j)
#pragma unroll
      for (int e = 0; e < 8; ++e) {
        int m = mbase + i * 16 + e + 8 * half;   // output channel
        int n = nbase + j * 16 + r;              // token
        int t = n >> 11, s = n & 2047;
        size_t xi = ((size_t)t * CH + m) * SP + s;
        y[xi] = acc[i][j][e] + bias[m] + xres[xi];
      }
}

// ---------------- launch ----------------
extern "C" void kernel_launch(void* const* d_in, const int* in_sizes, int n_in,
                              void* d_out, int out_size, void* d_ws, size_t ws_size,
                              hipStream_t stream) {
  (void)in_sizes; (void)n_in; (void)out_size; (void)ws_size;

  const float* x      = (const float*)d_in[0];
  const float* gnw    = (const float*)d_in[1];
  const float* w_in   = (const float*)d_in[2];   // (2304,768)
  const float* b_in   = (const float*)d_in[3];
  const float* q_sc   = (const float*)d_in[4];
  const float* q_bi   = (const float*)d_in[5];
  const float* k_sc   = (const float*)d_in[6];
  const float* k_bi   = (const float*)d_in[7];
  const float* reltab = (const float*)d_in[8];   // (32,12)
  const float* w_out  = (const float*)d_in[9];   // (768,768)
  const float* b_out  = (const float*)d_in[10];
  float* y = (float*)d_out;

  char* ws = (char*)d_ws;
  size_t off = 0;
  auto take = [&](size_t bytes) -> char* {
    char* p = ws + off;
    off += (bytes + 255) & ~(size_t)255;
    return p;
  };
  u16*   xbt   = (u16*)  take((size_t)NTOK * CH * sizeof(u16));          // 50 MB
  u16*   wi_bf = (u16*)  take((size_t)M_QKV * CH * sizeof(u16));         // 3.5 MB
  u16*   wo_bf = (u16*)  take((size_t)CH * CH * sizeof(u16));            // 1.2 MB
  float* rstd  = (float*)take((size_t)TT * HEADS * sizeof(float));
  float* bmat  = (float*)take((size_t)HEADS * TT * TT * sizeof(float));
  float* qkvb  = (float*)take((size_t)3 * HEADS * SP * TT * HD * sizeof(float)); // 302 MB
  u16*   obt   = (u16*)  take((size_t)NTOK * CH * sizeof(u16));          // 50 MB

  // 1) group RMS stats
  gn_stats_kernel<<<TT * HEADS, 256, 0, stream>>>(x, rstd);
  // 2) normalize + transpose to token-major bf16
  norm_transpose_kernel<<<dim3(SP / 32, CH / 32, TT), dim3(32, 8), 0, stream>>>(x, rstd, gnw, xbt);
  // 3) convert weights to bf16
  {
    int n1 = M_QKV * CH;
    cvt_bf16_kernel<<<(n1 + 255) / 256, 256, 0, stream>>>(w_in, wi_bf, n1);
    int n2 = CH * CH;
    cvt_bf16_kernel<<<(n2 + 255) / 256, 256, 0, stream>>>(w_out, wo_bf, n2);
  }
  // 4) relative position bias table
  bias_table_kernel<<<(HEADS * TT * TT + 255) / 256, 256, 0, stream>>>(reltab, bmat);
  // 5) QKV GEMM: 2304 x 32768 x 768
  {
    int tiles = (M_QKV / 32) * (NTOK / 32);   // 73728
    gemm_qkv_kernel<<<tiles / 8, 256, 0, stream>>>(wi_bf, xbt, b_in, qkvb);
  }
  // 6) attention: one wave per (head, s)
  attn_kernel<<<(HEADS * SP) / 4, 128, 0, stream>>>(qkvb, q_sc, q_bi, k_sc, k_bi, bmat, obt);
  // 7) output GEMM + bias + residual
  {
    int tiles = (CH / 32) * (NTOK / 32);      // 24576
    gemm_out_kernel<<<tiles / 8, 256, 0, stream>>>(wo_bf, obt, b_out, x, y);
  }
}
